// LSTMVAE_51101520888501
// MI455X (gfx1250) — matrix operand order repaired and checked by reading, not compiled, not run on hardware
//
#include <hip/hip_runtime.h>
#include <hip/hip_bf16.h>
#include <math.h>

// ---------------------------------------------------------------------------
// LSTM-VAE on MI455X (gfx1250).
//   B=64, T=256, D=1024, L=2, G=128, 4D=4096
// GEMMs run as v_wmma_f32_16x16x32_bf16 (fp32 accum); states/cells in fp32.
// Weights spectral-normed + converted to bf16 once (67MB, L2-resident).
// x converted to bf16 once up front so the serial step kernels are pure bf16.
// ---------------------------------------------------------------------------

typedef __attribute__((ext_vector_type(16))) __bf16 v16bf;
typedef __attribute__((ext_vector_type(8)))  float  v8f;

#define BB 64
#define TT 256
#define DD 1024
#define LL 2
#define GG 128
#define FD 4096   // 4*D

// ---- helpers ---------------------------------------------------------------

__device__ __forceinline__ unsigned short f2bf(float f) {
  unsigned u = __builtin_bit_cast(unsigned, f);
  u += 0x7FFFu + ((u >> 16) & 1u);            // round-to-nearest-even
  return (unsigned short)(u >> 16);
}

struct U4Pack { uint4 a, b; };

// 16 bf16 from two 16B-aligned chunks
__device__ __forceinline__ v16bf load16bf(const unsigned short* p0,
                                          const unsigned short* p1) {
  U4Pack u;
  u.a = *(const uint4*)p0;
  u.b = *(const uint4*)p1;
  return __builtin_bit_cast(v16bf, u);
}

// A fragment (ISA 7.12.2, 16-bit A 16x32):
//   lanes 0-15 : K = 0..7 (v0-3), 16..23 (v4-7)
//   lanes 16-31: K = 8..15,       24..31
__device__ __forceinline__ v16bf loadA_bf(const unsigned short* rowp, int kb, int hi) {
  return load16bf(rowp + kb + hi * 8, rowp + kb + 16 + hi * 8);
}

// B fragment: column n = lane&15; lanes 0-15 K=0..7,16..23, lanes 16-31 K=8..15,24..31.
// W stored row-major [4096,1024]; B[k,n] = W[n,k] -> contiguous in k.
__device__ __forceinline__ v16bf loadB(const unsigned short* p) {
  return load16bf(p, p + 16);
}

__device__ __forceinline__ v8f wmma_bf16(v16bf a, v16bf b, v8f c) {
  return __builtin_amdgcn_wmma_f32_16x16x32_bf16(false, a, false, b,
                                                 (short)0, c, false, false);
}

__device__ __forceinline__ float sigm(float x) { return 1.0f / (1.0f + __expf(-x)); }

__device__ __forceinline__ float block_reduce_sum(float v) {
  __shared__ float red[256];
  int tid = threadIdx.x;
  red[tid] = v; __syncthreads();
  for (int s = 128; s > 0; s >>= 1) {
    if (tid < s) red[tid] += red[tid + s];
    __syncthreads();
  }
  return red[0];
}

// ---- fused LSTM step: gates GEMM (WMMA) + cell nonlinearity ----------------
// grid = 256 blocks (4 batch-row tiles x 64 d-slices), 128 threads (4 waves).
// Wave w computes gate w's 16x16 tile; LDS collects i,f,g,o; cell fused after.
// Two accumulators (x-path, h-path) break the WMMA D->C dependency chain.
__global__ __launch_bounds__(128) void lstm_step(
    const unsigned short* __restrict__ Ain,   // bf16 input rows
    long aStride,                             // elements between batch rows
    const unsigned short* __restrict__ Wih,   // [4096,1024] bf16 (SN'd)
    const unsigned short* __restrict__ Whh,   // [4096,1024] bf16 (SN'd)
    const float* __restrict__ bias,           // [4096]
    const unsigned short* __restrict__ Hin,   // [64,1024] bf16 (prev h)
    unsigned short* __restrict__ Hout,        // [64,1024] bf16 (new h)
    float* __restrict__ Cst,                  // [64,1024] f32 c state (in/out)
    float* __restrict__ Hf32,                 // optional f32 h mirror
    float* __restrict__ outp,                 // optional output row base
    long outStride,                           // out row stride (elements)
    unsigned short* __restrict__ xout)        // optional bf16 h mirror (dec fb)
{
  __shared__ float ldsG[4][16][16];

  const int tid  = threadIdx.x;
  const int gate = tid >> 5;          // wave id = gate (i,f,g,o)
  const int lane = tid & 31;
  const int row  = lane & 15;
  const int hi   = lane >> 4;
  const int brow = blockIdx.x & 3;    // batch tile (16 rows)
  const int dblk = blockIdx.x >> 2;   // d slice (16 cols of D)

  const int aRow = brow * 16 + row;                   // global batch row
  const int ncol = gate * 1024 + dblk * 16 + row;     // gate-space column

  const unsigned short* arow = Ain + (size_t)aRow * aStride;
  const unsigned short* hrow = Hin + (size_t)aRow * DD;
  const unsigned short* wB0  = Wih + (size_t)ncol * DD;
  const unsigned short* wB1  = Whh + (size_t)ncol * DD;

  v8f acc0 = {};
  v8f acc1 = {};

#pragma unroll 4
  for (int kb = 0; kb < DD; kb += 32) {
    __builtin_prefetch(wB0 + kb + 512, 0, 1);   // speculative; OOB is dropped
    __builtin_prefetch(wB1 + kb + 512, 0, 1);
    v16bf a0 = loadA_bf(arow, kb, hi);
    v16bf b0 = loadB(wB0 + kb + hi * 8);
    v16bf a1 = loadA_bf(hrow, kb, hi);
    v16bf b1 = loadB(wB1 + kb + hi * 8);
    acc0 = wmma_bf16(a0, b0, acc0);
    acc1 = wmma_bf16(a1, b1, acc1);
  }

  // C layout: acc[r] -> row m = hi*8 + r, col n = lane&15
  const float bb = bias[ncol];
#pragma unroll
  for (int r = 0; r < 8; ++r)
    ldsG[gate][hi * 8 + r][row] = acc0[r] + acc1[r] + bb;

  __syncthreads();

  // fused cell update for this block's 16x16 (batch x d) slice
#pragma unroll
  for (int e = tid; e < 256; e += 128) {
    const int bl = e >> 4, dd = e & 15;
    const int bg = brow * 16 + bl;
    const int dg = dblk * 16 + dd;
    const size_t idx = (size_t)bg * DD + dg;
    const float ig = ldsG[0][bl][dd];
    const float fg = ldsG[1][bl][dd];
    const float gg = ldsG[2][bl][dd];
    const float og = ldsG[3][bl][dd];
    const float cn = sigm(fg) * Cst[idx] + sigm(ig) * tanhf(gg);
    Cst[idx] = cn;
    const float h = sigm(og) * tanhf(cn);
    Hout[idx] = f2bf(h);
    if (Hf32) Hf32[idx] = h;
    if (xout) xout[idx] = f2bf(h);
    if (outp) outp[(size_t)bg * outStride + dg] = h;
  }
}

// ---- bulk f32 -> bf16 conversion (x mirror) --------------------------------
__global__ void conv_f32_bf16(const float* __restrict__ src,
                              unsigned short* __restrict__ dst) {
  const size_t i = ((size_t)blockIdx.x * 256 + threadIdx.x) * 4;
  const float4 f = *(const float4*)(src + i);
  ushort4 o;
  o.x = f2bf(f.x); o.y = f2bf(f.y); o.z = f2bf(f.z); o.w = f2bf(f.w);
  *(ushort4*)(dst + i) = o;
}

// ---- spectral norm pipeline ------------------------------------------------

struct SNArgs {
  const float*    W[8];    // [4096,1024] each
  const float*    u[8];    // [4096]
  unsigned short* dst[8];  // bf16 [4096,1024]
};

__global__ void sn_norm_u(SNArgs a, float* __restrict__ scal) {
  const int m = blockIdx.x;
  const float* u = a.u[m];
  float s = 0.f;
  for (int i = threadIdx.x; i < FD; i += 256) { float v = u[i]; s += v * v; }
  s = block_reduce_sum(s);
  if (threadIdx.x == 0) scal[m * 4 + 0] = sqrtf(s);
}

__global__ void sn_v(SNArgs a, const float* __restrict__ scal, float* __restrict__ v) {
  const int m = blockIdx.x >> 2;
  const int d = (blockIdx.x & 3) * 256 + threadIdx.x;
  const float* W = a.W[m];
  const float* u = a.u[m];
  const float inv = 1.0f / (scal[m * 4 + 0] + 1e-12f);
  float acc = 0.f;
  for (int r = 0; r < FD; ++r) acc += W[(size_t)r * DD + d] * u[r];
  v[m * DD + d] = acc * inv;
}

__global__ void sn_norm_v(const float* __restrict__ v, float* __restrict__ scal) {
  const int m = blockIdx.x;
  float s = 0.f;
  for (int i = threadIdx.x; i < DD; i += 256) { float x = v[m * DD + i]; s += x * x; }
  s = block_reduce_sum(s);
  if (threadIdx.x == 0) scal[m * 4 + 1] = sqrtf(s);
}

__global__ void sn_t(SNArgs a, const float* __restrict__ scal,
                     const float* __restrict__ v, float* __restrict__ t) {
  const int m = blockIdx.x >> 4;
  const int r = (blockIdx.x & 15) * 256 + threadIdx.x;
  const float* W = a.W[m];
  const float inv = 1.0f / (scal[m * 4 + 1] + 1e-12f);
  float acc = 0.f;
  for (int d = 0; d < DD; ++d) acc += W[(size_t)r * DD + d] * v[m * DD + d];
  t[m * FD + r] = acc * inv;
}

__global__ void sn_sigma(const float* __restrict__ t, float* __restrict__ scal) {
  const int m = blockIdx.x;
  float s = 0.f;
  for (int i = threadIdx.x; i < FD; i += 256) { float x = t[m * FD + i]; s += x * x; }
  s = block_reduce_sum(s);
  if (threadIdx.x == 0) {
    const float n = sqrtf(s);
    scal[m * 4 + 2] = (n + 1e-12f) / s;   // 1/sigma, sigma = u2.(Wv) = |Wv|
  }
}

__global__ void sn_scale(SNArgs a, const float* __restrict__ scal) {
  const int m = blockIdx.x >> 14;
  const size_t e = ((size_t)(blockIdx.x & 16383)) * 256 + threadIdx.x;
  a.dst[m][e] = f2bf(a.W[m][e] * scal[m * 4 + 2]);
}

// ---- VAE head --------------------------------------------------------------

__global__ void vae_musig(
    const float* __restrict__ hstate, const float* __restrict__ cstate, // [L,B,D]
    const float* __restrict__ h_muW, const float* __restrict__ h_mub,
    const float* __restrict__ h_sgW, const float* __restrict__ h_sgb,
    const float* __restrict__ c_muW, const float* __restrict__ c_mub,
    const float* __restrict__ c_sgW, const float* __restrict__ c_sgb,
    const float* __restrict__ eps_h, const float* __restrict__ eps_c,
    float* __restrict__ z, float* __restrict__ muO, float* __restrict__ sgO) {
  const int gid = blockIdx.x * 256 + threadIdx.x;       // 32768 threads
  const int g  = gid & 127;
  const int b  = (gid >> 7) & 63;
  const int l  = (gid >> 13) & 1;
  const int hc = gid >> 14;
  const float* s   = (hc ? cstate : hstate) + ((size_t)l * BB + b) * DD;
  const float* muW = (hc ? c_muW : h_muW) + ((size_t)l * GG + g) * DD;
  const float* sgW = (hc ? c_sgW : h_sgW) + ((size_t)l * GG + g) * DD;
  float mu = (hc ? c_mub : h_mub)[l * GG + g];
  float sg = (hc ? c_sgb : h_sgb)[l * GG + g];
  for (int d = 0; d < DD; ++d) { const float x = s[d]; mu += x * muW[d]; sg += x * sgW[d]; }
  const int oi = ((hc * LL + l) * BB + b) * GG + g;
  muO[oi] = mu;
  sgO[oi] = sg;
  const float e = (hc ? eps_c : eps_h)[((size_t)l * BB + b) * GG + g];
  z[oi] = mu + e * __expf(sg);
}

__global__ void vae_out(
    const float* __restrict__ z,
    const float* __restrict__ h_outW, const float* __restrict__ h_outb,
    const float* __restrict__ c_outW, const float* __restrict__ c_outb,
    unsigned short* __restrict__ dec_h0,  // bf16 [L,B,D] (decoder init h, parity 0)
    float* __restrict__ dec_c) {          // f32  [L,B,D]
  const int gid = blockIdx.x * 256 + threadIdx.x;       // 262144 threads
  const int d  = gid & 1023;
  const int b  = (gid >> 10) & 63;
  const int l  = (gid >> 16) & 1;
  const int hc = gid >> 17;
  const float* zz = z + ((size_t)(hc * LL + l) * BB + b) * GG;
  const float* W  = (hc ? c_outW : h_outW) + ((size_t)l * DD + d) * GG;
  float acc = (hc ? c_outb : h_outb)[l * DD + d];
  for (int g = 0; g < GG; ++g) acc += zz[g] * W[g];
  const size_t idx = ((size_t)l * BB + b) * DD + d;
  if (hc) dec_c[idx] = acc; else dec_h0[idx] = f2bf(acc);
}

// ---- misc ------------------------------------------------------------------

__global__ void zero_state(unsigned short* __restrict__ hbf0, float* __restrict__ c0) {
  const int i = blockIdx.x * 256 + threadIdx.x;         // 131072 = L*B*D
  hbf0[i] = 0;
  c0[i] = 0.f;
}

__global__ void init_xbuf(const float* __restrict__ initial, unsigned short* __restrict__ xbuf) {
  const int i = blockIdx.x * 256 + threadIdx.x;         // 65536 = B*D
  xbuf[i] = f2bf(initial[i & 1023]);
}

// ---------------------------------------------------------------------------

extern "C" void kernel_launch(void* const* d_in, const int* in_sizes, int n_in,
                              void* d_out, int out_size, void* d_ws, size_t ws_size,
                              hipStream_t stream) {
  (void)in_sizes; (void)n_in; (void)out_size; (void)ws_size;

  const float* x        = (const float*)d_in[0];
  const float* initial  = (const float*)d_in[1];
  const float* enc_Wih  = (const float*)d_in[2];
  const float* enc_Whh  = (const float*)d_in[3];
  const float* enc_b    = (const float*)d_in[4];
  const float* enc_u_ih = (const float*)d_in[5];
  const float* enc_u_hh = (const float*)d_in[6];
  const float* dec_Wih  = (const float*)d_in[7];
  const float* dec_Whh  = (const float*)d_in[8];
  const float* dec_b    = (const float*)d_in[9];
  const float* dec_u_ih = (const float*)d_in[10];
  const float* dec_u_hh = (const float*)d_in[11];
  const float* h_muW = (const float*)d_in[12]; const float* h_mub = (const float*)d_in[13];
  const float* h_sgW = (const float*)d_in[14]; const float* h_sgb = (const float*)d_in[15];
  const float* h_outW = (const float*)d_in[16]; const float* h_outb = (const float*)d_in[17];
  const float* c_muW = (const float*)d_in[18]; const float* c_mub = (const float*)d_in[19];
  const float* c_sgW = (const float*)d_in[20]; const float* c_sgb = (const float*)d_in[21];
  const float* c_outW = (const float*)d_in[22]; const float* c_outb = (const float*)d_in[23];
  const float* eps_h = (const float*)d_in[24]; const float* eps_c = (const float*)d_in[25];

  // workspace carve-up
  char* base = (char*)d_ws;
  size_t off = 0;
  auto take = [&](size_t n) -> void* {
    void* p = base + off;
    off = (off + n + 255) & ~(size_t)255;
    return p;
  };
  unsigned short* wbf[4];
  for (int i = 0; i < 4; ++i) wbf[i] = (unsigned short*)take((size_t)LL * FD * DD * 2);
  unsigned short* xbf    = (unsigned short*)take((size_t)BB * TT * DD * 2);   // bf16 x mirror
  unsigned short* enc_hbf = (unsigned short*)take((size_t)2 * LL * BB * DD * 2); // parity x [L,B,D]
  unsigned short* dec_hbf = (unsigned short*)take((size_t)2 * LL * BB * DD * 2);
  float* enc_c  = (float*)take((size_t)LL * BB * DD * 4);
  float* dec_c  = (float*)take((size_t)LL * BB * DD * 4);
  float* enc_hf = (float*)take((size_t)LL * BB * DD * 4);
  unsigned short* xbuf = (unsigned short*)take((size_t)BB * DD * 2);
  float* zbuf = (float*)take((size_t)2 * LL * BB * GG * 4);
  float* snv  = (float*)take((size_t)8 * DD * 4);
  float* snt  = (float*)take((size_t)8 * FD * 4);
  float* scal = (float*)take((size_t)32 * 4);

  SNArgs sa;
  {
    const float* Wm[4] = {enc_Wih, enc_Whh, dec_Wih, dec_Whh};
    const float* Um[4] = {enc_u_ih, enc_u_hh, dec_u_ih, dec_u_hh};
    for (int m = 0; m < 8; ++m) {
      sa.W[m]   = Wm[m >> 1] + (size_t)(m & 1) * FD * DD;
      sa.u[m]   = Um[m >> 1] + (size_t)(m & 1) * FD;
      sa.dst[m] = wbf[m >> 1] + (size_t)(m & 1) * FD * DD;
    }
  }

  const size_t LBD = (size_t)LL * BB * DD;
  const size_t BD  = (size_t)BB * DD;

  // 1) init state, bf16 x mirror, spectral norm -> bf16 weights
  zero_state<<<512, 256, 0, stream>>>(enc_hbf, enc_c);
  conv_f32_bf16<<<16384, 256, 0, stream>>>(x, xbf);   // 16.7M elems, 4/thread
  sn_norm_u<<<8,   256, 0, stream>>>(sa, scal);
  sn_v     <<<32,  256, 0, stream>>>(sa, scal, snv);
  sn_norm_v<<<8,   256, 0, stream>>>(snv, scal);
  sn_t     <<<128, 256, 0, stream>>>(sa, scal, snv, snt);
  sn_sigma <<<8,   256, 0, stream>>>(snt, scal);
  sn_scale <<<131072, 256, 0, stream>>>(sa, scal);

  // 2) encoder: wavefront order (layer0 then layer1 each step)
  for (int t = 0; t < TT; ++t) {
    const size_t pi = (size_t)(t & 1) * LBD;
    const size_t po = (size_t)((t + 1) & 1) * LBD;
    lstm_step<<<256, 128, 0, stream>>>(
        xbf + (size_t)t * DD, (long)((size_t)TT * DD),
        wbf[0], wbf[1], enc_b,
        enc_hbf + pi, enc_hbf + po, enc_c, enc_hf,
        nullptr, 0, nullptr);
    lstm_step<<<256, 128, 0, stream>>>(
        enc_hbf + po, (long)DD,
        wbf[0] + (size_t)FD * DD, wbf[1] + (size_t)FD * DD, enc_b + FD,
        enc_hbf + pi + BD, enc_hbf + po + BD, enc_c + BD, enc_hf + BD,
        nullptr, 0, nullptr);
  }

  // 3) VAE head: mu/sg (to d_out) + z, then z -> decoder initial states
  float* outF = (float*)d_out;
  float* muO  = outF + (size_t)BB * TT * DD;
  float* sgO  = muO + (size_t)2 * LL * BB * GG;
  vae_musig<<<128, 256, 0, stream>>>(enc_hf, enc_c,
                                     h_muW, h_mub, h_sgW, h_sgb,
                                     c_muW, c_mub, c_sgW, c_sgb,
                                     eps_h, eps_c, zbuf, muO, sgO);
  vae_out<<<1024, 256, 0, stream>>>(zbuf, h_outW, h_outb, c_outW, c_outb,
                                    dec_hbf /*parity 0*/, dec_c);
  init_xbuf<<<256, 256, 0, stream>>>(initial, xbuf);

  // 4) autoregressive decoder; top-layer h -> out row + bf16 feedback
  for (int t = 0; t < TT; ++t) {
    const size_t pi = (size_t)(t & 1) * LBD;
    const size_t po = (size_t)((t + 1) & 1) * LBD;
    lstm_step<<<256, 128, 0, stream>>>(
        xbuf, (long)DD,
        wbf[2], wbf[3], dec_b,
        dec_hbf + pi, dec_hbf + po, dec_c, nullptr,
        nullptr, 0, nullptr);
    lstm_step<<<256, 128, 0, stream>>>(
        dec_hbf + po, (long)DD,
        wbf[2] + (size_t)FD * DD, wbf[3] + (size_t)FD * DD, dec_b + FD,
        dec_hbf + pi + BD, dec_hbf + po + BD, dec_c + BD, nullptr,
        outF + (size_t)t * DD, (long)((size_t)TT * DD), xbuf);
  }
}
